// MixtralMoE_9019431321896
// MI455X (gfx1250) — compile-verified, matrix operand order checked by reading, not run on hardware
//
#include <hip/hip_runtime.h>
#include <hip/hip_bf16.h>

// ---------------------------------------------------------------------------
// MoE (Mixtral-style) for gfx1250: router + top-2 gather + bf16 WMMA SwiGLU
// T=8192 tokens, H=2048 hidden, F=7168 intermediate, E=8 experts, top_k=2
// Double-buffered LDS, async global->LDS for bf16 A tiles, fp32->bf16 weight
// conversion fused into the B-tile staging.
// ---------------------------------------------------------------------------

constexpr int kT = 8192;
constexpr int kH = 2048;
constexpr int kF = 7168;
constexpr int kE = 8;

constexpr int BM = 128;   // token-tile
constexpr int BN = 128;   // output-col tile
constexpr int BK = 32;    // K step (WMMA bf16 K)
constexpr int LDSTR = 40; // bf16 elems per LDS row (80B: 16B aligned, 20-dword stride)

typedef __attribute__((ext_vector_type(16))) __bf16 v16bf;
typedef __attribute__((ext_vector_type(8)))  float  v8f;
typedef __attribute__((ext_vector_type(4)))  int    v4i;

union FragBF { v16bf v; unsigned int u[8]; };

__device__ __forceinline__ __bf16 f2bf(float f) {
  unsigned int u = __float_as_uint(f);
  unsigned int r = (u + 0x7FFFu + ((u >> 16) & 1u)) >> 16;
  unsigned short s = (unsigned short)r;
  union { unsigned short s; __bf16 b; } cvt;
  cvt.s = s;
  return cvt.b;
}

// 16-byte async global->LDS copy (ASYNCcnt-tracked on CDNA5).
// Toolchain prototype expects generic v4i* operands (per compiler diagnostic):
// (global_src, lds_dst, imm_offset, imm_cpol).
__device__ __forceinline__ void async_copy16(void* lds_dst, const void* g_src) {
#if __has_builtin(__builtin_amdgcn_global_load_async_to_lds_b128)
  __builtin_amdgcn_global_load_async_to_lds_b128((v4i*)g_src, (v4i*)lds_dst, 0, 0);
#else
  *(uint4*)lds_dst = *(const uint4*)g_src;
#endif
}

__device__ __forceinline__ void wait_async0() {
#if __has_builtin(__builtin_amdgcn_s_wait_asynccnt)
  __builtin_amdgcn_s_wait_asynccnt(0);
#else
  asm volatile("s_wait_asynccnt 0" ::: "memory");
#endif
}

// Load a 16x32 (A) or [n][k]-stored B bf16 fragment from an LDS tile,
// row-major with row stride LDSTR bf16. Lane mapping per ISA 7.12.2.
__device__ __forceinline__ v16bf load_frag(const __bf16* base, int row0, int lane) {
  FragBF f;
  int r = row0 + (lane & 15);
  int g = lane >> 4;
  const unsigned int* p = (const unsigned int*)(base + r * LDSTR);
#pragma unroll
  for (int v = 0; v < 8; ++v) {
    int k = ((v < 4) ? 0 : 16) + g * 8 + 2 * (v & 3);
    f.u[v] = p[k >> 1];
  }
  return f.v;
}

// ---------------------------------------------------------------------------
// 0) zero output + expert counters
// ---------------------------------------------------------------------------
__global__ __launch_bounds__(256) void moe_zero_kernel(float* __restrict__ out,
                                                       int* __restrict__ cnt) {
  size_t i = (size_t)blockIdx.x * blockDim.x + threadIdx.x;
  ((float4*)out)[i] = make_float4(0.f, 0.f, 0.f, 0.f);
  if (blockIdx.x == 0 && threadIdx.x < kE) cnt[threadIdx.x] = 0;
}

// ---------------------------------------------------------------------------
// 1) fp32 -> bf16 conversion of hidden states
// ---------------------------------------------------------------------------
__global__ __launch_bounds__(256) void moe_cvt_kernel(const float* __restrict__ X,
                                                      __bf16* __restrict__ Xb) {
  size_t i = ((size_t)blockIdx.x * blockDim.x + threadIdx.x) * 4;
  float4 v = *(const float4*)(X + i);
  union { unsigned short s[4]; uint2 u; } p;
  union { unsigned short s; __bf16 b; } c;
  c.b = f2bf(v.x); p.s[0] = c.s;
  c.b = f2bf(v.y); p.s[1] = c.s;
  c.b = f2bf(v.z); p.s[2] = c.s;
  c.b = f2bf(v.w); p.s[3] = c.s;
  *(uint2*)(Xb + i) = p.u;
}

// ---------------------------------------------------------------------------
// 2) router: logits -> softmax -> top-2 -> renormalize -> compacted lists
// ---------------------------------------------------------------------------
__global__ __launch_bounds__(256) void moe_router_kernel(
    const float* __restrict__ X, const float* __restrict__ Wg,
    int* __restrict__ cnt, int* __restrict__ tok_list, float* __restrict__ w_list) {
  __shared__ float red[kE * 256];
  int t = blockIdx.x;
  const float* x = X + (size_t)t * kH;
  float acc[kE];
#pragma unroll
  for (int e = 0; e < kE; ++e) acc[e] = 0.f;
  for (int h = threadIdx.x; h < kH; h += 256) {
    float xv = x[h];
    const float* wgr = Wg + (size_t)h * kE;
#pragma unroll
    for (int e = 0; e < kE; ++e) acc[e] += xv * wgr[e];
  }
#pragma unroll
  for (int e = 0; e < kE; ++e) red[e * 256 + threadIdx.x] = acc[e];
  __syncthreads();
  for (int s = 128; s > 0; s >>= 1) {
    if (threadIdx.x < s) {
#pragma unroll
      for (int e = 0; e < kE; ++e)
        red[e * 256 + threadIdx.x] += red[e * 256 + threadIdx.x + s];
    }
    __syncthreads();
  }
  if (threadIdx.x == 0) {
    float lg[kE], p[kE];
    float mx = -1e30f;
#pragma unroll
    for (int e = 0; e < kE; ++e) { lg[e] = red[e * 256]; mx = fmaxf(mx, lg[e]); }
#pragma unroll
    for (int e = 0; e < kE; ++e) p[e] = __expf(lg[e] - mx);
    int i1 = 0; float v1 = p[0];
#pragma unroll
    for (int e = 1; e < kE; ++e) if (p[e] > v1) { v1 = p[e]; i1 = e; }
    int i2 = -1; float v2 = -1.f;
#pragma unroll
    for (int e = 0; e < kE; ++e) if (e != i1 && p[e] > v2) { v2 = p[e]; i2 = e; }
    float inv = 1.f / (v1 + v2);           // softmax denominator cancels in renorm
    int s1 = atomicAdd(&cnt[i1], 1);
    tok_list[i1 * kT + s1] = t; w_list[i1 * kT + s1] = v1 * inv;
    int s2 = atomicAdd(&cnt[i2], 1);
    tok_list[i2 * kT + s2] = t; w_list[i2 * kT + s2] = v2 * inv;
  }
}

// ---------------------------------------------------------------------------
// 3) up-projection per expert: act[m,:] = silu(x W1) * (x W3), bf16 out
//    gathered token rows; dual-GEMM sharing the A tile; double-buffered
// ---------------------------------------------------------------------------
__global__ __launch_bounds__(256) void moe_up_kernel(
    const __bf16* __restrict__ Xb, const float* __restrict__ W1e,
    const float* __restrict__ W3e, const int* __restrict__ cnt,
    const int* __restrict__ tok_list, __bf16* __restrict__ act, int e) {
  __shared__ __bf16 sA[2][BM * LDSTR];
  __shared__ __bf16 sB1[2][BN * LDSTR];
  __shared__ __bf16 sB3[2][BN * LDSTR];
  __shared__ int sTok[BM];

  int n_e = cnt[e];
  int mBase = blockIdx.y * BM;
  if (mBase >= n_e) return;
  int nBase = blockIdx.x * BN;

  if (threadIdx.x < BM) {
    int m = mBase + threadIdx.x;
    sTok[threadIdx.x] = (m < n_e) ? tok_list[e * kT + m] : -1;
  }
  __syncthreads();

  auto loadA = [&](int buf, int k0) {
#pragma unroll
    for (int c = threadIdx.x; c < BM * 4; c += 256) {   // 512 x 16B chunks
      int m = c >> 2, seg = c & 3;
      int tokv = sTok[m];
      if (tokv < 0) tokv = 0;                           // dummy row; discarded later
      async_copy16(&sA[buf][m * LDSTR + seg * 8],
                   Xb + (size_t)tokv * kH + k0 + seg * 8);
    }
  };
  auto loadB = [&](int buf, int k0) {
#pragma unroll
    for (int idx = threadIdx.x; idx < (BK * BN) / 4; idx += 256) {
      int k  = idx >> 5;          // 0..31
      int n4 = (idx & 31) * 4;    // 0..124
      size_t g = (size_t)(k0 + k) * kF + (nBase + n4);
      float4 a = *(const float4*)(W1e + g);
      float4 b = *(const float4*)(W3e + g);
      __bf16* d1 = &sB1[buf][n4 * LDSTR + k];
      d1[0] = f2bf(a.x); d1[LDSTR] = f2bf(a.y);
      d1[2 * LDSTR] = f2bf(a.z); d1[3 * LDSTR] = f2bf(a.w);
      __bf16* d3 = &sB3[buf][n4 * LDSTR + k];
      d3[0] = f2bf(b.x); d3[LDSTR] = f2bf(b.y);
      d3[2 * LDSTR] = f2bf(b.z); d3[3 * LDSTR] = f2bf(b.w);
    }
  };

  int lane  = threadIdx.x & 31;
  int wid   = threadIdx.x >> 5;
  int waveM = (wid & 3) * 32;   // 4 M-strips of 32
  int waveN = (wid >> 2) * 64;  // 2 N-strips of 64

  v8f acc1[2][4] = {};
  v8f acc3[2][4] = {};

  loadA(0, 0);
  loadB(0, 0);
  wait_async0();
  __syncthreads();

  for (int k0 = 0; k0 < kH; k0 += BK) {
    int cur = (k0 / BK) & 1, nxt = cur ^ 1;
    if (k0 + BK < kH) { loadA(nxt, k0 + BK); loadB(nxt, k0 + BK); }

    v16bf fa[2], fb1[4], fb3[4];
#pragma unroll
    for (int mi = 0; mi < 2; ++mi) fa[mi] = load_frag(sA[cur], waveM + mi * 16, lane);
#pragma unroll
    for (int ni = 0; ni < 4; ++ni) {
      fb1[ni] = load_frag(sB1[cur], waveN + ni * 16, lane);
      fb3[ni] = load_frag(sB3[cur], waveN + ni * 16, lane);
    }
#pragma unroll
    for (int mi = 0; mi < 2; ++mi)
#pragma unroll
      for (int ni = 0; ni < 4; ++ni) {
        acc1[mi][ni] = __builtin_amdgcn_wmma_f32_16x16x32_bf16(
            false, fa[mi], false, fb1[ni], (short)0, acc1[mi][ni], false, false);
        acc3[mi][ni] = __builtin_amdgcn_wmma_f32_16x16x32_bf16(
            false, fa[mi], false, fb3[ni], (short)0, acc3[mi][ni], false, false);
      }

    wait_async0();
    __syncthreads();
  }

  // epilogue: silu(a1) * a3 -> bf16 intermediate
  int g = lane >> 4, nl = lane & 15;
#pragma unroll
  for (int mi = 0; mi < 2; ++mi)
#pragma unroll
    for (int ni = 0; ni < 4; ++ni) {
      int n = nBase + waveN + ni * 16 + nl;
#pragma unroll
      for (int v = 0; v < 8; ++v) {
        int m = mBase + waveM + mi * 16 + v + 8 * g;
        if (m < n_e) {
          float a1 = acc1[mi][ni][v], a3 = acc3[mi][ni][v];
          float gated = (a1 / (1.f + __expf(-a1))) * a3;
          act[(size_t)m * kF + n] = f2bf(gated);
        }
      }
    }
}

// ---------------------------------------------------------------------------
// 4) down-projection per expert: out[token,:] += w * (act W2); double-buffered
// ---------------------------------------------------------------------------
__global__ __launch_bounds__(256) void moe_down_kernel(
    const __bf16* __restrict__ act, const float* __restrict__ W2e,
    const int* __restrict__ cnt, const int* __restrict__ tok_list,
    const float* __restrict__ w_list, float* __restrict__ out, int e) {
  __shared__ __bf16 sA[2][BM * LDSTR];
  __shared__ __bf16 sB[2][BN * LDSTR];
  __shared__ int   sTok[BM];
  __shared__ float sW[BM];

  int n_e = cnt[e];
  int mBase = blockIdx.y * BM;
  if (mBase >= n_e) return;
  int nBase = blockIdx.x * BN;

  if (threadIdx.x < BM) {
    int m = mBase + threadIdx.x;
    sTok[threadIdx.x] = (m < n_e) ? tok_list[e * kT + m] : -1;
    sW[threadIdx.x]   = (m < n_e) ? w_list[e * kT + m] : 0.f;
  }
  __syncthreads();

  auto loadA = [&](int buf, int k0) {
#pragma unroll
    for (int c = threadIdx.x; c < BM * 4; c += 256) {
      int m = c >> 2, seg = c & 3;
      async_copy16(&sA[buf][m * LDSTR + seg * 8],
                   act + (size_t)(mBase + m) * kF + k0 + seg * 8);
    }
  };
  auto loadB = [&](int buf, int k0) {
#pragma unroll
    for (int idx = threadIdx.x; idx < (BK * BN) / 4; idx += 256) {
      int k  = idx >> 5;
      int n4 = (idx & 31) * 4;
      float4 a = *(const float4*)(W2e + (size_t)(k0 + k) * kH + (nBase + n4));
      __bf16* d = &sB[buf][n4 * LDSTR + k];
      d[0] = f2bf(a.x); d[LDSTR] = f2bf(a.y);
      d[2 * LDSTR] = f2bf(a.z); d[3 * LDSTR] = f2bf(a.w);
    }
  };

  int lane  = threadIdx.x & 31;
  int wid   = threadIdx.x >> 5;
  int waveM = (wid & 3) * 32;
  int waveN = (wid >> 2) * 64;

  v8f acc[2][4] = {};

  loadA(0, 0);
  loadB(0, 0);
  wait_async0();
  __syncthreads();

  for (int k0 = 0; k0 < kF; k0 += BK) {
    int cur = (k0 / BK) & 1, nxt = cur ^ 1;
    if (k0 + BK < kF) { loadA(nxt, k0 + BK); loadB(nxt, k0 + BK); }

    v16bf fa[2], fb[4];
#pragma unroll
    for (int mi = 0; mi < 2; ++mi) fa[mi] = load_frag(sA[cur], waveM + mi * 16, lane);
#pragma unroll
    for (int ni = 0; ni < 4; ++ni) fb[ni] = load_frag(sB[cur], waveN + ni * 16, lane);
#pragma unroll
    for (int mi = 0; mi < 2; ++mi)
#pragma unroll
      for (int ni = 0; ni < 4; ++ni)
        acc[mi][ni] = __builtin_amdgcn_wmma_f32_16x16x32_bf16(
            false, fa[mi], false, fb[ni], (short)0, acc[mi][ni], false, false);

    wait_async0();
    __syncthreads();
  }

  // epilogue: scaled scatter-add into fp32 output (expert kernels serialized)
  int g = lane >> 4, nl = lane & 15;
#pragma unroll
  for (int mi = 0; mi < 2; ++mi)
#pragma unroll
    for (int ni = 0; ni < 4; ++ni) {
      int n = nBase + waveN + ni * 16 + nl;
#pragma unroll
      for (int v = 0; v < 8; ++v) {
        int mL = waveM + mi * 16 + v + 8 * g;
        int tokv = sTok[mL];
        if (tokv >= 0) {
          float* o = out + (size_t)tokv * kH + n;
          *o += sW[mL] * acc[mi][ni][v];
        }
      }
    }
}

// ---------------------------------------------------------------------------
// host launcher
// ---------------------------------------------------------------------------
extern "C" void kernel_launch(void* const* d_in, const int* in_sizes, int n_in,
                              void* d_out, int out_size, void* d_ws, size_t ws_size,
                              hipStream_t stream) {
  (void)in_sizes; (void)n_in; (void)out_size; (void)ws_size;
  const float* X  = (const float*)d_in[0];
  const float* Wg = (const float*)d_in[1];
  const float* W1 = (const float*)d_in[2];
  const float* W2 = (const float*)d_in[3];
  const float* W3 = (const float*)d_in[4];
  float* out = (float*)d_out;

  // workspace layout (all 256B aligned)
  char* ws = (char*)d_ws;
  int*    cnt = (int*)ws;                                   // 256 B
  int*    tok = (int*)(ws + 256);                           // E*T*4
  float*  wts = (float*)(ws + 256 + (size_t)kE * kT * 4);   // E*T*4
  __bf16* Xb  = (__bf16*)(ws + 256 + 2 * (size_t)kE * kT * 4);
  __bf16* acti = (__bf16*)((char*)Xb + (size_t)kT * kH * 2);

  moe_zero_kernel<<<(kT * kH) / 1024, 256, 0, stream>>>(out, cnt);
  moe_cvt_kernel<<<(kT * kH) / 1024, 256, 0, stream>>>(X, Xb);
  moe_router_kernel<<<kT, 256, 0, stream>>>(X, Wg, cnt, tok, wts);

  for (int e = 0; e < kE; ++e) {
    const float* W1e = W1 + (size_t)e * kH * kF;
    const float* W3e = W3 + (size_t)e * kH * kF;
    const float* W2e = W2 + (size_t)e * kF * kH;
    moe_up_kernel<<<dim3(kF / BN, kT / BM), 256, 0, stream>>>(
        Xb, W1e, W3e, cnt, tok, acti, e);
    moe_down_kernel<<<dim3(kH / BN, kT / BM), 256, 0, stream>>>(
        acti, W2e, cnt, tok, wts, out, e);
  }
}
